// FSMNMultiHeadAttention_72395968741703
// MI455X (gfx1250) — compile-verified, matrix operand order checked by reading, not run on hardware
//
#include <hip/hip_runtime.h>
#include <cstdint>

typedef __attribute__((ext_vector_type(16))) __bf16 v16bf;
typedef __attribute__((ext_vector_type(8)))  float  v8f;
typedef __attribute__((ext_vector_type(8)))  __bf16 bf16x8;
typedef __attribute__((ext_vector_type(4)))  unsigned int u32x4;
typedef __attribute__((ext_vector_type(4)))  int i32x4;
typedef __attribute__((ext_vector_type(8)))  int i32x8;

constexpr int B_ = 4, T_ = 1024, C_ = 1024, H_ = 16, DH_ = 64, KC_ = 31, LEFT_ = 15;
constexpr int M_ = B_ * T_;                 // 4096 rows for projections
constexpr float QK_SCALE = 0.35355339059327373f;  // 64^-0.25, applied to q and k

#if defined(__has_builtin)
#if __has_builtin(__builtin_amdgcn_tensor_load_to_lds)
#define HAVE_TDM 1
#endif
#endif

__device__ __forceinline__ void tdm_issue(u32x4 g0, i32x8 g1) {
#if defined(HAVE_TDM)
  i32x4 z4 = {};
#if __clang_major__ >= 23
  i32x8 z8 = {};
  __builtin_amdgcn_tensor_load_to_lds(g0, g1, z4, z4, z8, 0);
#else
  __builtin_amdgcn_tensor_load_to_lds(g0, g1, z4, z4, 0);
#endif
#else
  (void)g0; (void)g1;
#endif
}

__device__ __forceinline__ void tdm_wait() {
#if defined(__has_builtin)
#if __has_builtin(__builtin_amdgcn_s_wait_tensorcnt)
  __builtin_amdgcn_s_wait_tensorcnt(0);
#endif
#endif
}

// ---- fragment loader: row-major, contiguous K; lane pattern K = (lane/16)*8 + (j<8?j:j+8)
__device__ __forceinline__ v16bf load_frag16(const __bf16* __restrict__ p) {
  bf16x8 lo = *(const bf16x8*)(p);
  bf16x8 hi = *(const bf16x8*)(p + 16);
  v16bf r;
#pragma unroll
  for (int i = 0; i < 8; ++i) { r[i] = lo[i]; r[i + 8] = hi[i]; }
  return r;
}

__global__ void cvt_bf16_kernel(const float* __restrict__ src, __bf16* __restrict__ dst, int n) {
  int i = blockIdx.x * blockDim.x + threadIdx.x;
  if (i < n) dst[i] = (__bf16)src[i];
}

// C[M,N] = A[M,K] @ W[N,K]^T (+bias[col]) (+addend[row*N+col]).
// 8 waves/block, one 16x16 tile each, all sharing the same 16 A-rows.
// A tile (16x32 bf16 per K-step) is staged into LDS by the Tensor Data Mover,
// double-buffered one K-step ahead; B is loaded directly (unique per wave).
__global__ void gemm_xwT_kernel(const __bf16* __restrict__ A, const __bf16* __restrict__ W,
                                const float* __restrict__ bias, const float* __restrict__ addend,
                                float* __restrict__ Cout, int M, int N, int K) {
  __shared__ __bf16 asmem[2][16 * 32];
  const int wave = threadIdx.x >> 5;
  const int lane = threadIdx.x & 31;
  const int tilesN = N >> 4;
  const int tile0 = blockIdx.x * 8;            // block covers 8 consecutive n-tiles, same m-row
  const int m0 = (tile0 / tilesN) << 4;
  const int n0 = ((tile0 % tilesN) + wave) << 4;
  const int l16 = lane & 15;
  const int kh = (lane >> 4) << 3;
  const __bf16* brow = W + (size_t)(n0 + l16) * K;

  // D# group 1 (constant): data_size=2B, tensor 32(k) x 16(rows), tile 32x16, row stride K
  i32x8 g1 = {};
  g1[0] = 0x10000;       // workgroup_mask=0, data_size=1 (2 bytes)
  g1[1] = 32 << 16;      // tensor_dim0[15:0]
  g1[2] = 16 << 16;      // tensor_dim0[31:16]=0 | tensor_dim1[15:0]
  g1[3] = 32 << 16;      // tensor_dim1[31:16]=0 | tile_dim0
  g1[4] = 16;            // tile_dim1 | tile_dim2=0
  g1[5] = K;             // tensor_dim0_stride[31:0]
  const uint32_t lds_base = (uint32_t)(uintptr_t)(void*)&asmem[0][0];

  auto stage = [&](int step) {
#if defined(HAVE_TDM)
    uint64_t gaddr = (uint64_t)(uintptr_t)(A + (size_t)m0 * K + (step << 5));
    u32x4 g0 = {};
    g0[0] = 1u;  // count=1 valid descriptor
    g0[1] = lds_base + (uint32_t)((step & 1) * (16 * 32 * 2));
    g0[2] = (uint32_t)gaddr;
    g0[3] = ((uint32_t)(gaddr >> 32) & 0x01FFFFFFu) | 0x80000000u;  // addr[56:32] | type=2
    tdm_issue(g0, g1);
#else
    for (int e = lane; e < 16 * 32; e += 32)
      asmem[step & 1][e] = A[(size_t)(m0 + (e >> 5)) * K + (step << 5) + (e & 31)];
#endif
  };

  if (wave == 0) { stage(0); tdm_wait(); }
  __syncthreads();

  v8f acc = {};
  const int steps = K >> 5;
  for (int i = 0; i < steps; ++i) {
    const int k0 = i << 5;
    if (wave == 0 && (i + 1) < steps) stage(i + 1);   // prefetch next A tile via TDM
    v16bf af = load_frag16(&asmem[i & 1][l16 * 32 + kh]);       // ds_load
    v16bf bfrag = load_frag16(brow + k0 + kh);                   // global_load_b128
    if ((i + 2) < steps) __builtin_prefetch(brow + k0 + 64, 0, 0);  // global_prefetch_b8
    acc = __builtin_amdgcn_wmma_f32_16x16x32_bf16(false, af, false, bfrag, (short)0, acc, false, false);
    if (wave == 0) tdm_wait();
    __syncthreads();
  }

  const int col = n0 + l16;
  const int rbase = m0 + ((lane >> 4) << 3);
  const float bc = bias ? bias[col] : 0.0f;
#pragma unroll
  for (int r = 0; r < 8; ++r) {
    const int row = rbase + r;
    float o = acc[r] + bc;
    if (addend) o += addend[(size_t)row * N + col];
    Cout[(size_t)row * N + col] = o;
  }
}

// RoPE + fold in dh^-0.25 scale; f32 in, bf16 out
__global__ void rope_scale_kernel(const float* __restrict__ src, const float* __restrict__ cosb,
                                  const float* __restrict__ sinb, __bf16* __restrict__ dst) {
  int i = blockIdx.x * blockDim.x + threadIdx.x;
  if (i >= M_ * C_) return;
  int c = i & (C_ - 1);
  int t = (i >> 10) & (T_ - 1);
  int d = c & (DH_ - 1);
  float x = src[i];
  float xr = (d < DH_ / 2) ? -src[i + DH_ / 2] : src[i - DH_ / 2];
  float v = x * cosb[t * DH_ + d] + xr * sinb[t * DH_ + d];
  dst[i] = (__bf16)(v * QK_SCALE);
}

// depthwise conv-31 FSMN memory + residual; also emit V transposed (b,c,s) as bf16
__global__ void fsmn_kernel(const float* __restrict__ v, const float* __restrict__ mpad,
                            const float* __restrict__ w, float* __restrict__ fsm,
                            __bf16* __restrict__ vt) {
  int i = blockIdx.x * blockDim.x + threadIdx.x;
  if (i >= M_ * C_) return;
  int c = i & (C_ - 1);
  int bt = i >> 10;
  int t = bt & (T_ - 1);
  int b = bt >> 10;
  float m = mpad[bt];
  float vm = v[i] * m;
  float acc = 0.0f;
#pragma unroll
  for (int j = 0; j < KC_; ++j) {
    int tt = t + j - LEFT_;
    if (tt >= 0 && tt < T_) {
      int bt2 = b * T_ + tt;
      acc += v[(size_t)bt2 * C_ + c] * mpad[bt2] * w[c * KC_ + j];
    }
  }
  fsm[i] = (acc + vm) * m;
  vt[((size_t)(b * C_ + c)) * T_ + t] = (__bf16)v[i];
}

// qk[bh,t,s] = q.k^T + mask ; wave computes a 16(t) x 64(s) strip
__global__ void attn_qk_kernel(const __bf16* __restrict__ q, const __bf16* __restrict__ k,
                               const float* __restrict__ mask, float* __restrict__ qkout) {
  const int lane = threadIdx.x & 31;
  constexpr int NSTRIP = T_ / 64;          // 16 strips of 64 cols
  constexpr int SPB = (T_ / 16) * NSTRIP;  // 1024 strips per (b,h)
  int strip = blockIdx.x * (blockDim.x >> 5) + (threadIdx.x >> 5);
  if (strip >= B_ * H_ * SPB) return;
  const int bh = strip / SPB;
  const int r = strip % SPB;
  const int m0 = (r / NSTRIP) << 4;
  const int n0 = (r % NSTRIP) << 6;
  const int b = bh / H_, h = bh % H_;
  const int l16 = lane & 15, kh = (lane >> 4) << 3;
  const __bf16* qbase = q + (size_t)b * T_ * C_ + h * DH_;
  const __bf16* kbase = k + (size_t)b * T_ * C_ + h * DH_;
  const __bf16* arow = qbase + (size_t)(m0 + l16) * C_;
  v8f acc[4] = {};
#pragma unroll
  for (int d0 = 0; d0 < DH_; d0 += 32) {
    v16bf af = load_frag16(arow + d0 + kh);
#pragma unroll
    for (int j = 0; j < 4; ++j) {
      v16bf bfrag = load_frag16(kbase + (size_t)(n0 + j * 16 + l16) * C_ + d0 + kh);
      acc[j] = __builtin_amdgcn_wmma_f32_16x16x32_bf16(false, af, false, bfrag, (short)0, acc[j], false, false);
    }
  }
  float* outb = qkout + (size_t)bh * T_ * T_;
  const int rbase = m0 + ((lane >> 4) << 3);
#pragma unroll
  for (int j = 0; j < 4; ++j) {
    const int col = n0 + j * 16 + l16;
#pragma unroll
    for (int rr = 0; rr < 8; ++rr) {
      const int row = rbase + rr;
      outb[(size_t)row * T_ + col] = acc[j][rr] + mask[(size_t)row * T_ + col];
    }
  }
}

// per-row softmax stats: (max, 1/sum(exp)); one 256-thread block per row
__global__ void softmax_stats_kernel(const float* __restrict__ qk, float2* __restrict__ stats) {
  __shared__ float red[16];
  const int row = blockIdx.x;
  const float4 v4 = ((const float4*)(qk + (size_t)row * T_))[threadIdx.x];
  float mx = fmaxf(fmaxf(v4.x, v4.y), fmaxf(v4.z, v4.w));
#pragma unroll
  for (int off = 16; off > 0; off >>= 1) mx = fmaxf(mx, __shfl_xor(mx, off, 32));
  const int wid = threadIdx.x >> 5, lane = threadIdx.x & 31;
  if (lane == 0) red[wid] = mx;
  __syncthreads();
  float mall = red[0];
#pragma unroll
  for (int i = 1; i < 8; ++i) mall = fmaxf(mall, red[i]);
  float s = __expf(v4.x - mall) + __expf(v4.y - mall) + __expf(v4.z - mall) + __expf(v4.w - mall);
#pragma unroll
  for (int off = 16; off > 0; off >>= 1) s += __shfl_xor(s, off, 32);
  if (lane == 0) red[8 + wid] = s;
  __syncthreads();
  if (threadIdx.x == 0) {
    float tot = 0.0f;
#pragma unroll
    for (int i = 0; i < 8; ++i) tot += red[8 + i];
    stats[row] = make_float2(mall, 1.0f / tot);
  }
}

// wv[t, h*64+d] = softmax(qk) @ v ; exp applied on-the-fly to A fragments.
__global__ void attn_wv_kernel(const float* __restrict__ qk, const float2* __restrict__ stats,
                               const __bf16* __restrict__ vt, __bf16* __restrict__ wv) {
  const int lane = threadIdx.x & 31;
  constexpr int SPB = T_ / 16;  // 64 strips per (b,h)
  int strip = blockIdx.x * (blockDim.x >> 5) + (threadIdx.x >> 5);
  if (strip >= B_ * H_ * SPB) return;
  const int bh = strip / SPB;
  const int m0 = (strip % SPB) << 4;
  const int b = bh / H_, h = bh % H_;
  const int l16 = lane & 15, kh = (lane >> 4) << 3;
  const int tr = m0 + l16;
  const float2 st = stats[bh * T_ + tr];
  const float mx = st.x, inv = st.y;
  const float* prow = qk + ((size_t)bh * T_ + tr) * T_;
  const __bf16* vbase = vt + (size_t)(b * C_ + h * DH_) * T_;
  v8f acc[4] = {};
  for (int s0 = 0; s0 < T_; s0 += 32) {
    const float* p = prow + s0 + kh;
    float4 p0 = *(const float4*)(p);
    float4 p1 = *(const float4*)(p + 4);
    float4 p2 = *(const float4*)(p + 16);
    float4 p3 = *(const float4*)(p + 20);
    float pf[16] = {p0.x, p0.y, p0.z, p0.w, p1.x, p1.y, p1.z, p1.w,
                    p2.x, p2.y, p2.z, p2.w, p3.x, p3.y, p3.z, p3.w};
    v16bf af;
#pragma unroll
    for (int i = 0; i < 16; ++i) af[i] = (__bf16)(__expf(pf[i] - mx) * inv);
#pragma unroll
    for (int j = 0; j < 4; ++j) {
      v16bf bfrag = load_frag16(vbase + (size_t)(j * 16 + l16) * T_ + s0 + kh);
      acc[j] = __builtin_amdgcn_wmma_f32_16x16x32_bf16(false, af, false, bfrag, (short)0, acc[j], false, false);
    }
  }
  const int rbase = m0 + ((lane >> 4) << 3);
#pragma unroll
  for (int j = 0; j < 4; ++j) {
    const int d = j * 16 + l16;
#pragma unroll
    for (int rr = 0; rr < 8; ++rr) {
      wv[((size_t)b * T_ + rbase + rr) * C_ + h * DH_ + d] = (__bf16)acc[j][rr];
    }
  }
}

extern "C" void kernel_launch(void* const* d_in, const int* in_sizes, int n_in,
                              void* d_out, int out_size, void* d_ws, size_t ws_size,
                              hipStream_t stream) {
  (void)in_sizes; (void)n_in; (void)out_size; (void)ws_size;
  const float* x    = (const float*)d_in[0];
  const float* mask = (const float*)d_in[1];
  const float* mpad = (const float*)d_in[2];
  const float* cosb = (const float*)d_in[3];
  const float* sinb = (const float*)d_in[4];
  const float* Wq   = (const float*)d_in[5];
  const float* bq   = (const float*)d_in[6];
  const float* Wk   = (const float*)d_in[7];
  const float* Wv   = (const float*)d_in[8];
  const float* bv   = (const float*)d_in[9];
  const float* Wo   = (const float*)d_in[10];
  const float* bo   = (const float*)d_in[11];
  const float* fw   = (const float*)d_in[12];

  float* out = (float*)d_out;                 // (B,T,C)
  float* qk  = out + (size_t)M_ * C_;         // (B,H,T,T) — second tuple output

  char* ws = (char*)d_ws;
  auto take = [&](size_t bytes) { char* p = ws; ws += (bytes + 255) & ~(size_t)255; return p; };
  __bf16* xb   = (__bf16*)take((size_t)M_ * C_ * 2);
  __bf16* Wqb  = (__bf16*)take((size_t)C_ * C_ * 2);
  __bf16* Wkb  = (__bf16*)take((size_t)C_ * C_ * 2);
  __bf16* Wvb  = (__bf16*)take((size_t)C_ * C_ * 2);
  __bf16* Wob  = (__bf16*)take((size_t)C_ * C_ * 2);
  float*  qraw = (float*)take((size_t)M_ * C_ * 4);
  float*  kraw = (float*)take((size_t)M_ * C_ * 4);
  float*  vraw = (float*)take((size_t)M_ * C_ * 4);
  __bf16* qb   = (__bf16*)take((size_t)M_ * C_ * 2);
  __bf16* kb   = (__bf16*)take((size_t)M_ * C_ * 2);
  __bf16* vt   = (__bf16*)take((size_t)M_ * C_ * 2);
  float*  fsm  = (float*)take((size_t)M_ * C_ * 4);
  float2* stats= (float2*)take((size_t)B_ * H_ * T_ * sizeof(float2));
  __bf16* wvb  = (__bf16*)take((size_t)M_ * C_ * 2);

  const int elems = M_ * C_;      // 4M
  const int welems = C_ * C_;     // 1M
  cvt_bf16_kernel<<<(elems + 255) / 256, 256, 0, stream>>>(x, xb, elems);
  cvt_bf16_kernel<<<(welems + 255) / 256, 256, 0, stream>>>(Wq, Wqb, welems);
  cvt_bf16_kernel<<<(welems + 255) / 256, 256, 0, stream>>>(Wk, Wkb, welems);
  cvt_bf16_kernel<<<(welems + 255) / 256, 256, 0, stream>>>(Wv, Wvb, welems);
  cvt_bf16_kernel<<<(welems + 255) / 256, 256, 0, stream>>>(Wo, Wob, welems);

  const int gemmBlocks = ((M_ / 16) * (C_ / 16)) / 8;   // 8 waves/block
  gemm_xwT_kernel<<<gemmBlocks, 256, 0, stream>>>(xb, Wqb, bq, nullptr, qraw, M_, C_, C_);
  gemm_xwT_kernel<<<gemmBlocks, 256, 0, stream>>>(xb, Wkb, nullptr, nullptr, kraw, M_, C_, C_);
  gemm_xwT_kernel<<<gemmBlocks, 256, 0, stream>>>(xb, Wvb, bv, nullptr, vraw, M_, C_, C_);

  rope_scale_kernel<<<(elems + 255) / 256, 256, 0, stream>>>(qraw, cosb, sinb, qb);
  rope_scale_kernel<<<(elems + 255) / 256, 256, 0, stream>>>(kraw, cosb, sinb, kb);
  fsmn_kernel<<<(elems + 255) / 256, 256, 0, stream>>>(vraw, mpad, fw, fsm, vt);

  attn_qk_kernel<<<(B_ * H_ * 1024) / 8, 256, 0, stream>>>(qb, kb, mask, qk);
  softmax_stats_kernel<<<B_ * H_ * T_, 256, 0, stream>>>(qk, stats);
  attn_wv_kernel<<<(B_ * H_ * (T_ / 16)) / 8, 256, 0, stream>>>(qk, stats, vt, wvb);

  gemm_xwT_kernel<<<gemmBlocks, 256, 0, stream>>>(wvb, Wob, bo, fsm, out, M_, C_, C_);
}
